// RNN_vanilla_188978561251
// MI455X (gfx1250) — compile-verified, matrix operand order checked by reading, not run on hardware
//
#include <hip/hip_runtime.h>
#include <hip/hip_bf16.h>

// ---------------- CDNA5 vector types (probe-verified signatures) ----------
typedef __attribute__((ext_vector_type(16))) __bf16 v16bf;
typedef __attribute__((ext_vector_type(8)))  float  v8f;
typedef __attribute__((ext_vector_type(4)))  int    i4;

union Pack16 { i4 q[2]; v16bf v; };   // 32B: two b128 loads -> one WMMA operand

// ---------------- problem constants ---------------------------------------
#define T_STEPS   16384
#define HDIM      2048
#define WROW      (HDIM + 2)          // W_i2h row stride (2050)
#define NWG       64                  // persistent workgroups (1 per WGP)
#define ROWS_PER_WG 32                // 2048 / 64
#define NTHREADS  128                 // 4 waves (wave32)

// LDS layout (dynamic shared, ~161 KB of the 320 KB WGP LDS)
#define ROW_STRIDE_E 2056                              // padded bf16 elems/row
#define ROW_STRIDE_B (ROW_STRIDE_E * 2)                // 4112 B (16B aligned, %256==16)
#define WB_BYTES  (ROWS_PER_WG * ROW_STRIDE_B)         // 131584
#define W2O_OFF   WB_BYTES
#define W2O_BYTES (3 * HDIM * 4)                       // 24576
#define WIN_OFF   (W2O_OFF + W2O_BYTES)
#define WIN_BYTES (3 * ROWS_PER_WG * 4)                // 384
#define RED_OFF   (WIN_OFF + WIN_BYTES)
#define RED_BYTES (2 * 4 * 16 * 4)                     // 512
#define HB_OFF    (RED_OFF + RED_BYTES)                // h stage buffer (bf16)
#define HB_BYTES  (HDIM * 2)                           // 4096
#define SMEM_BYTES (HB_OFF + HB_BYTES)                 // 161152

#define MAGIC 0x13579BDFu

__device__ __forceinline__ float softplus_f(float a) {
    float m = fmaxf(a, 0.0f);
    return m + log1pf(__expf(-fabsf(a)));   // stable jax.nn.softplus
}

// low 32 bits of a generic LDS pointer == group-segment byte offset
__device__ __forceinline__ unsigned lds_off(const void* p) {
    return (unsigned)(size_t)p;
}

__global__ __launch_bounds__(NTHREADS, 1) void rnn_persist(
    const float* __restrict__ x,        // (T, 2)
    const float* __restrict__ W,        // (H, 2050)
    const float* __restrict__ bi,       // (H)
    const float* __restrict__ Wo,       // (3, H)
    const float* __restrict__ bo,       // (3)
    const float* __restrict__ noise,    // (T, H)
    float* __restrict__ out,            // [T*3 | T*H]
    unsigned* __restrict__ cnt,         // grid-barrier counter (d_ws)
    unsigned* __restrict__ flag,        // bootstrap flag (d_ws)
    __hip_bfloat16* __restrict__ hb0,   // h double buffer, bf16 (d_ws, L2-resident)
    __hip_bfloat16* __restrict__ hb1)
{
    extern __shared__ char smem[];
    __hip_bfloat16* wB  = (__hip_bfloat16*)smem;          // [32][2056] bf16 W^T tiles
    float*          w2o = (float*)(smem + W2O_OFF);       // [3][2048] (WG0 only)
    float*          win = (float*)(smem + WIN_OFF);       // w0[32], w1[32], b[32]
    float*          red = (float*)(smem + RED_OFF);       // [2 blk][4 wave][16]
    char*           hls = smem + HB_OFF;                  // staged h (bf16, 4KB)

    const int tid  = threadIdx.x;
    const int wg   = blockIdx.x;
    const int wave = __builtin_amdgcn_readfirstlane(tid) >> 5;   // SGPR-uniform
    const int lane = tid & 31;
    const int sel  = (lane & 16) ? 16 : 0;     // K-half byte select (A & B layouts)

    float* outVal = out;
    float* outHid = out + (size_t)3 * T_STEPS;

    // ---------------- preload: weights -> LDS, bootstrap barrier ----------
    if (wg == 0 && tid == 0)
        __hip_atomic_store(cnt, 0u, __ATOMIC_RELAXED, __HIP_MEMORY_SCOPE_AGENT);

    for (int idx = tid; idx < ROWS_PER_WG * HDIM; idx += NTHREADS) {
        int r = idx >> 11, k = idx & (HDIM - 1);
        float w = W[(size_t)(wg * ROWS_PER_WG + r) * WROW + 2 + k];
        wB[r * ROW_STRIDE_E + k] = __float2bfloat16(w);
    }
    if (tid < ROWS_PER_WG) {
        int rg = wg * ROWS_PER_WG + tid;
        win[tid]      = W[(size_t)rg * WROW + 0];
        win[32 + tid] = W[(size_t)rg * WROW + 1];
        win[64 + tid] = bi[rg];
    }
    if (wg == 0) {
        for (int idx = tid; idx < 3 * HDIM; idx += NTHREADS) w2o[idx] = Wo[idx];
        for (int idx = tid; idx < HDIM; idx += NTHREADS)
            hb0[idx] = __float2bfloat16(0.0f);           // h_{-1} = 0
    }
    __syncthreads();

    if (wg == 0) {
        if (tid == 0) {
            __threadfence();
            __hip_atomic_store(flag, MAGIC, __ATOMIC_RELEASE, __HIP_MEMORY_SCOPE_AGENT);
        }
    } else {
        if (tid == 0) {
            while (__hip_atomic_load(flag, __ATOMIC_ACQUIRE, __HIP_MEMORY_SCOPE_AGENT) != MAGIC)
                __builtin_amdgcn_s_sleep(2);
        }
        __syncthreads();
    }

    // per-wave B addressing: lane n -> column n of B = row n of this WG's W slab
    const char*    wBb    = (const char*)wB;
    const int      rsel0  = ((lane & 15)     ) * ROW_STRIDE_B + sel;  // row block 0
    const int      rsel1  = ((lane & 15) + 16) * ROW_STRIDE_B + sel;  // row block 1
    const unsigned hbdst  = lds_off(hls) + (unsigned)tid * 32u;       // stage dest
    const int      hgsrc  = tid * 32;                                 // stage src off
    const int      cbase  = wave * 64;      // this wave's first chunk (byte offset)

    // ---------------- time loop -------------------------------------------
    for (int t = 0; t < T_STEPS; ++t) {
        const __hip_bfloat16* hprev = (t & 1) ? hb1 : hb0;
        __hip_bfloat16*       hnext = (t & 1) ? hb0 : hb1;

        // stage h_{t-1} (4KB bf16) global -> LDS via CDNA5 async DMA path
        {
            unsigned long long gbase = (unsigned long long)hprev;
            asm volatile("global_load_async_to_lds_b128 %0, %1, %2 offset:0"
                         :: "v"(hbdst), "v"(hgsrc), "s"(gbase) : "memory");
            asm volatile("global_load_async_to_lds_b128 %0, %1, %2 offset:16"
                         :: "v"(hbdst), "v"(hgsrc), "s"(gbase) : "memory");
        }

        // WG0: readout for step t-1 (overlaps the async stage + others' GEMV)
        if (wg == 0 && t > 0 && tid < 96) {
            const float* hp = outHid + (size_t)(t - 1) * HDIM;
            int o = tid >> 5, l = tid & 31;
            float s = 0.0f;
            for (int k = l; k < HDIM; k += 32) s += w2o[o * HDIM + k] * hp[k];
            for (int off = 16; off > 0; off >>= 1) s += __shfl_xor(s, off, 32);
            if (l == 0) {
                float v = s + bo[o];
                outVal[(size_t)(t - 1) * 3 + o] = fminf(fmaxf(v, -1000.0f), 1000.0f);
            }
        }

        asm volatile("s_wait_asynccnt 0x0" ::: "memory");
        __syncthreads();                       // staged h visible to all waves

        // WMMA GEMV: A = h replicated across M rows, B = dense W^T tile.
        // 4 accumulators (even/odd chunks x 2 row blocks) halve the dep chain;
        // explicit 2-stage software pipeline keeps LDS loads ahead of WMMAs.
        v8f a0e = {}, a0o = {}, a1e = {}, a1o = {};
        Pack16 A[2], B0[2], B1[2];
        {
            const int cb = cbase;
            A[0].q[0]  = *(const i4*)(hls + cb + sel);
            A[0].q[1]  = *(const i4*)(hls + cb + sel + 32);
            B0[0].q[0] = *(const i4*)(wBb + rsel0 + cb);
            B0[0].q[1] = *(const i4*)(wBb + rsel0 + cb + 32);
            B1[0].q[0] = *(const i4*)(wBb + rsel1 + cb);
            B1[0].q[1] = *(const i4*)(wBb + rsel1 + cb + 32);
        }
#pragma unroll
        for (int i = 0; i < 16; ++i) {
            const int cur = i & 1, nxt = cur ^ 1;
            if (i < 15) {                      // prefetch chunk i+1 operands
                const int cb = cbase + (i + 1) * 256;
                A[nxt].q[0]  = *(const i4*)(hls + cb + sel);
                A[nxt].q[1]  = *(const i4*)(hls + cb + sel + 32);
                B0[nxt].q[0] = *(const i4*)(wBb + rsel0 + cb);
                B0[nxt].q[1] = *(const i4*)(wBb + rsel0 + cb + 32);
                B1[nxt].q[0] = *(const i4*)(wBb + rsel1 + cb);
                B1[nxt].q[1] = *(const i4*)(wBb + rsel1 + cb + 32);
            }
            if (i & 1) {
                a0o = __builtin_amdgcn_wmma_f32_16x16x32_bf16(false, A[cur].v, false, B0[cur].v,
                                                              (short)0, a0o, false, false);
                a1o = __builtin_amdgcn_wmma_f32_16x16x32_bf16(false, A[cur].v, false, B1[cur].v,
                                                              (short)0, a1o, false, false);
            } else {
                a0e = __builtin_amdgcn_wmma_f32_16x16x32_bf16(false, A[cur].v, false, B0[cur].v,
                                                              (short)0, a0e, false, false);
                a1e = __builtin_amdgcn_wmma_f32_16x16x32_bf16(false, A[cur].v, false, B1[cur].v,
                                                              (short)0, a1e, false, false);
            }
        }
        // D rows identical (A replicated); lane n, VGPR0 = result[col n]
        if (lane < 16) {
            red[(0 * 4 + wave) * 16 + lane] = a0e[0] + a0o[0];
            red[(1 * 4 + wave) * 16 + lane] = a1e[0] + a1o[0];
        }
        __syncthreads();

        if (tid < ROWS_PER_WG) {
            int blk = tid >> 4, n = tid & 15;
            float a = red[(blk * 4 + 0) * 16 + n] + red[(blk * 4 + 1) * 16 + n]
                    + red[(blk * 4 + 2) * 16 + n] + red[(blk * 4 + 3) * 16 + n];
            int rg = wg * ROWS_PER_WG + tid;
            float x0 = x[2 * t], x1 = x[2 * t + 1];
            a += win[tid] * x0 + win[32 + tid] * x1 + win[64 + tid]
               + noise[(size_t)t * HDIM + rg];
            float hn = softplus_f(a);
            outHid[(size_t)t * HDIM + rg] = hn;          // hidden_all output (fp32)
            hnext[rg] = __float2bfloat16(hn);            // next-step WMMA operand
            if (t + 1 < T_STEPS)
                __builtin_prefetch(&noise[(size_t)(t + 1) * HDIM + rg], 0, 1);
        }
        __syncthreads();          // red[] reads done; hnext/outHid writes issued

        // grid barrier: arrive + monotonic-counter wait (L2 atomic)
        if (tid == 0) {
            __threadfence();
            __hip_atomic_fetch_add(cnt, 1u, __ATOMIC_ACQ_REL, __HIP_MEMORY_SCOPE_AGENT);
            const unsigned target = (unsigned)(t + 1) * NWG;
            while (__hip_atomic_load(cnt, __ATOMIC_ACQUIRE, __HIP_MEMORY_SCOPE_AGENT) < target)
                __builtin_amdgcn_s_sleep(1);
        }
        __syncthreads();
    }

    // ---------------- epilogue: out[T-1], deterministic ws reset ----------
    __syncthreads();
    if (tid == 0) {
        __threadfence();
        __hip_atomic_fetch_add(cnt, 1u, __ATOMIC_ACQ_REL, __HIP_MEMORY_SCOPE_AGENT);
    }
    if (wg == 0) {
        if (tid < 96) {
            const float* hp = outHid + (size_t)(T_STEPS - 1) * HDIM;
            int o = tid >> 5, l = tid & 31;
            float s = 0.0f;
            for (int k = l; k < HDIM; k += 32) s += w2o[o * HDIM + k] * hp[k];
            for (int off = 16; off > 0; off >>= 1) s += __shfl_xor(s, off, 32);
            if (l == 0) {
                float v = s + bo[o];
                outVal[(size_t)(T_STEPS - 1) * 3 + o] = fminf(fmaxf(v, -1000.0f), 1000.0f);
            }
        }
        __syncthreads();
        if (tid == 0) {
            const unsigned target = (unsigned)T_STEPS * NWG + NWG;  // all WGs fully done
            while (__hip_atomic_load(cnt, __ATOMIC_ACQUIRE, __HIP_MEMORY_SCOPE_AGENT) < target)
                __builtin_amdgcn_s_sleep(1);
            __hip_atomic_store(flag, 0u, __ATOMIC_RELEASE, __HIP_MEMORY_SCOPE_AGENT);
        }
    }
}

extern "C" void kernel_launch(void* const* d_in, const int* in_sizes, int n_in,
                              void* d_out, int out_size, void* d_ws, size_t ws_size,
                              hipStream_t stream) {
    (void)in_sizes; (void)n_in; (void)out_size; (void)ws_size;
    // setup_inputs() order:
    // 0: input_tensor (1,T,2) f32   1: ends   2: b   3: W_i2h (H,2050) f32
    // 4: b_i2h (H) f32   5: W_h2o (3,H) f32   6: b_h2o (3) f32   7: noise (T,H) f32
    const float* x  = (const float*)d_in[0];
    const float* W  = (const float*)d_in[3];
    const float* bi = (const float*)d_in[4];
    const float* Wo = (const float*)d_in[5];
    const float* bo = (const float*)d_in[6];
    const float* nz = (const float*)d_in[7];
    float* out = (float*)d_out;

    char* ws = (char*)d_ws;
    unsigned*       cnt  = (unsigned*)ws;
    unsigned*       flag = (unsigned*)(ws + 64);
    __hip_bfloat16* hb0  = (__hip_bfloat16*)(ws + 256);
    __hip_bfloat16* hb1  = (__hip_bfloat16*)(ws + 256 + HDIM * 2);

    hipLaunchKernelGGL(rnn_persist, dim3(NWG), dim3(NTHREADS), SMEM_BYTES, stream,
                       x, W, bi, Wo, bo, nz, out, cnt, flag, hb0, hb1);
}
